// Z2GraphAttention_83073257439361
// MI455X (gfx1250) — matrix operand order repaired and checked by reading, not compiled, not run on hardware
//
#include <hip/hip_runtime.h>
#include <hip/hip_bf16.h>
#include <math.h>

typedef _Float16 f16;
typedef __attribute__((ext_vector_type(16))) _Float16 v16h;
typedef __attribute__((ext_vector_type(8)))  _Float16 v8h;
typedef __attribute__((ext_vector_type(8)))  float    v8f;
typedef __attribute__((ext_vector_type(4)))  float    v4f;
typedef __attribute__((ext_vector_type(4)))  unsigned int u32x4;
typedef __attribute__((ext_vector_type(8)))  int      i32x8;
typedef __attribute__((ext_vector_type(4)))  int      i32x4;

#define N_TOK 8192
#define DMODEL 256
#define NHEAD 4
#define HDIM 64
#define KCHUNK 32
#define NCHUNK (N_TOK / KCHUNK)
// softmax scale (1/sqrt(64)) * log2(e), folded into Wq/bq so exp2 can be used
#define QSCALE (0.125f * 1.4426950408889634f)

static __device__ __forceinline__ v8f wmma16(v16h a, v16h b, v8f c) {
  return __builtin_amdgcn_wmma_f32_16x16x32_f16(false, a, false, b, (short)0, c,
                                                false, false);
}

// A fragment (16x32 f16): lane(l&15)=row m, 16 contiguous elems at k0+16s (s=l>>4)
static __device__ __forceinline__ v16h load_a16(const f16* base, int ld, int m0, int k0) {
  int l = threadIdx.x & 31;
  int s = l >> 4;
  const f16* p = base + (m0 + (l & 15)) * ld + k0 + 16 * s;
  union { v16h v; v8h h[2]; } u;
  u.h[0] = *(const v8h*)p;
  u.h[1] = *(const v8h*)(p + 8);
  return u.v;
}

// B fragment (32x16 f16) from "B-transposed" row-major storage (rows = n):
// lane(l&15)=col n, 16 contiguous elems at k0+16s
static __device__ __forceinline__ v16h load_b16(const f16* baseT, int ld, int k0, int n0) {
  int l = threadIdx.x & 31;
  int s = l >> 4;
  const f16* p = baseT + (n0 + (l & 15)) * ld + k0 + 16 * s;
  return *(const v16h*)p;
}

static __device__ __forceinline__ v8f zero8() {
  v8f z;
#pragma unroll
  for (int i = 0; i < 8; ++i) z[i] = 0.0f;
  return z;
}

// ---- Tensor Data Mover: 2-D tile load Global -> LDS (D# per ISA ch.8) ------
// data units are 8 bytes (data_size code 3).  tile_d1==1 gives a 1-D copy.
// 6-arg builtin form: (g0, g1, g2, g3, g4, cpol); groups 2-4 zero for <=2-D.
static __device__ __forceinline__ void tdm_load_2d(const void* gsrc, unsigned lds_off,
                                                   unsigned tile_d0, unsigned tile_d1,
                                                   unsigned long long stride_d0) {
  unsigned long long ga = (unsigned long long)(uintptr_t)gsrc;
  u32x4 g0;
  g0[0] = 1u;                                   // count=1, user mode, no gather
  g0[1] = lds_off;                              // LDS byte address
  g0[2] = (unsigned)ga;                         // global addr [31:0]
  g0[3] = (unsigned)((ga >> 32) & 0x1FFFFFFu) | (2u << 30);  // addr[56:32], type=2
  i32x8 g1;
  g1[0] = (int)(3u << 16);                      // wg_mask=0, data_size=8B, no flags
  g1[1] = (int)((tile_d0 & 0xFFFFu) << 16);     // tensor_dim0 = tile_d0 (no OOB clip)
  g1[2] = (int)(((tile_d0 >> 16) & 0xFFFFu) | ((tile_d1 & 0xFFFFu) << 16)); // dim0 hi | dim1 lo
  g1[3] = (int)(((tile_d1 >> 16) & 0xFFFFu) | ((tile_d0 & 0xFFFFu) << 16)); // dim1 hi | tile_dim0
  g1[4] = (int)(tile_d1 & 0xFFFFu);             // tile_dim1, tile_dim2=0 (2-D)
  g1[5] = (int)(stride_d0 & 0xFFFFFFFFu);       // tensor_dim0_stride lo
  g1[6] = (int)((stride_d0 >> 32) & 0xFFFFu);   // stride hi | dim1_stride lo = 0
  g1[7] = 0;
  i32x4 z4;
  z4[0] = z4[1] = z4[2] = z4[3] = 0;            // groups 2/3 unused (<=2-D)
  i32x8 z8;
#pragma unroll
  for (int i = 0; i < 8; ++i) z8[i] = 0;
  __builtin_amdgcn_tensor_load_to_lds(g0, g1, z4, z4, z8, 0);
}

// ---------------- prep: x -> f16 ----------------
__global__ void k_cvt_x(const float* __restrict__ x, f16* __restrict__ xh) {
  int i = (blockIdx.x * blockDim.x + threadIdx.x) * 8;
  v4f a = *(const v4f*)(x + i);
  v4f b = *(const v4f*)(x + i + 4);
  v8h o;
#pragma unroll
  for (int j = 0; j < 4; ++j) { o[j] = (f16)a[j]; o[4 + j] = (f16)b[j]; }
  *(v8h*)(xh + i) = o;
}

// ---------------- prep: transpose weights to f16, fold scale into Wq ---------
__global__ void k_prep_w(const float* __restrict__ Wq, const float* __restrict__ Wk,
                         const float* __restrict__ Wv, const float* __restrict__ Wo,
                         f16* __restrict__ WqT, f16* __restrict__ WkT,
                         f16* __restrict__ WvT, f16* __restrict__ WoT) {
  int sel = blockIdx.z;
  const float* W = (sel == 0) ? Wq : (sel == 1) ? Wk : (sel == 2) ? Wv : Wo;
  f16* WT = (sel == 0) ? WqT : (sel == 1) ? WkT : (sel == 2) ? WvT : WoT;
  float sc = (sel == 0) ? QSCALE : 1.0f;
  int n = blockIdx.y * 16 + threadIdx.y;  // output dim
  int k = blockIdx.x * 16 + threadIdx.x;  // input dim
  WT[n * DMODEL + k] = (f16)(W[k * DMODEL + n] * sc);
}

// ---------------- phase 1: QKV projection ----------------
// one wave: 16 rows x 128 cols (8 n-tiles, 64 acc VGPRs) of one of {Q, K, V}.
// Smaller acc footprint frees registers for a 2-deep rotating B-fragment
// buffer + A-next prefetch so loads stay ahead of the WMMAs.
__global__ void __launch_bounds__(128) k_proj_qkv(
    const f16* __restrict__ xh,
    const f16* __restrict__ WqT, const f16* __restrict__ WkT, const f16* __restrict__ WvT,
    const float* __restrict__ bq, const float* __restrict__ bk, const float* __restrict__ bv,
    f16* __restrict__ Qh, f16* __restrict__ Kh, f16* __restrict__ VTh) {
  int wid = threadIdx.x >> 5;
  int m0 = (blockIdx.x * 2 + (wid >> 1)) * 16;  // two row-tiles per block
  int nbase = (wid & 1) * 128;                  // column half per wave
  int sel = blockIdx.y;
  const f16* WT = (sel == 0) ? WqT : (sel == 1) ? WkT : WvT;
  const float* bias = (sel == 0) ? bq : (sel == 1) ? bk : bv;
  float bscale = (sel == 0) ? QSCALE : 1.0f;

  v8f acc[8];
#pragma unroll
  for (int t = 0; t < 8; ++t) acc[t] = zero8();

  v16h a = load_a16(xh, DMODEL, m0, 0);
  for (int k0 = 0; k0 < DMODEL; k0 += 32) {
    v16h an = (k0 + 32 < DMODEL) ? load_a16(xh, DMODEL, m0, k0 + 32) : a;
    v16h b0 = load_b16(WT, DMODEL, k0, nbase);
    v16h b1 = load_b16(WT, DMODEL, k0, nbase + 16);
#pragma unroll
    for (int nt = 0; nt < 8; ++nt) {
      v16h bn = (nt + 2 < 8) ? load_b16(WT, DMODEL, k0, nbase + (nt + 2) * 16) : b0;
      acc[nt] = wmma16(a, b0, acc[nt]);
      b0 = b1;
      b1 = bn;
    }
    a = an;
  }

  int l = threadIdx.x & 31;
  int s = l >> 4;
#pragma unroll
  for (int nt = 0; nt < 8; ++nt) {
    int n = nbase + nt * 16 + (l & 15);
    float b = bias[n] * bscale;
    if (sel == 2) {
      v8h o;
#pragma unroll
      for (int r = 0; r < 8; ++r) o[r] = (f16)(acc[nt][r] + b);
      *(v8h*)(VTh + n * N_TOK + m0 + 8 * s) = o;   // transposed, coalesced
    } else {
      f16* dst = (sel == 0) ? Qh : Kh;
#pragma unroll
      for (int r = 0; r < 8; ++r)
        dst[(m0 + r + 8 * s) * DMODEL + n] = (f16)(acc[nt][r] + b);
    }
  }
}

// ---------------- phase 2: flash attention with TDM double buffering --------
// block = 4 waves = 4 heads on one 16-query tile.  Wave 0 streams the 32-key
// chunk of K (16 KB contiguous) and the matching V^T slab (256x32, strided)
// into LDS with tensor_load_to_lds, double-buffered; s_wait_tensorcnt(2)
// retires the current chunk while the next one is in flight.  All eight LDS
// fragments (4x K, 4x V^T) are loaded right after the barrier so the V^T
// ds_loads overlap the S^T WMMAs and the softmax VALU work.
__global__ void __launch_bounds__(128) k_attn(
    const f16* __restrict__ Qh, const f16* __restrict__ Kh,
    const f16* __restrict__ VTh, f16* __restrict__ AOh) {
  __shared__ __align__(16) f16 sK[2][KCHUNK * DMODEL];    // 2 x 16 KB
  __shared__ __align__(16) f16 sVT[2][DMODEL * KCHUNK];   // 2 x 16 KB

  int h = threadIdx.x >> 5;     // 4 waves/block -> one head each
  int q0 = blockIdx.x * 16;
  int l = threadIdx.x & 31;
  int s = l >> 4;
  int d0 = h * HDIM;
  bool loader = (h == 0);       // wave 0 issues TDM (EXEC ignored, per-wave op)

  // resident Q B-fragments (Q^T in B position), scale already folded in
  v16h qb0 = load_b16(Qh, DMODEL, d0, q0);
  v16h qb1 = load_b16(Qh, DMODEL, d0 + 32, q0);

  if (loader) {
    tdm_load_2d(Kh, (unsigned)(uintptr_t)&sK[0][0],
                (KCHUNK * DMODEL * 2) / 8, 1, (KCHUNK * DMODEL * 2) / 8);
    tdm_load_2d(VTh, (unsigned)(uintptr_t)&sVT[0][0],
                (KCHUNK * 2) / 8, DMODEL, (N_TOK * 2) / 8);
  }

  v8f oacc[4];
#pragma unroll
  for (int t = 0; t < 4; ++t) oacc[t] = zero8();
  float mrow = -INFINITY;
  float lrow = 0.0f;

  for (int it = 0; it < NCHUNK; ++it) {
    int cur = it & 1;
    if (loader) {
      if (it + 1 < NCHUNK) {
        int j1 = (it + 1) * KCHUNK;
        tdm_load_2d(Kh + j1 * DMODEL, (unsigned)(uintptr_t)&sK[cur ^ 1][0],
                    (KCHUNK * DMODEL * 2) / 8, 1, (KCHUNK * DMODEL * 2) / 8);
        tdm_load_2d(VTh + j1, (unsigned)(uintptr_t)&sVT[cur ^ 1][0],
                    (KCHUNK * 2) / 8, DMODEL, (N_TOK * 2) / 8);
        __builtin_amdgcn_s_wait_tensorcnt(2);  // oldest pair (cur chunk) done
      } else {
        __builtin_amdgcn_s_wait_tensorcnt(0);
      }
    }
    __syncthreads();            // publish LDS chunk to all 4 heads

    const f16* kc = &sK[cur][0];
    const f16* vc = &sVT[cur][0];

    // issue ALL fragment loads up front: K frags feed the S^T WMMAs now,
    // V^T frags are consumed after the softmax, overlapping it.
    v16h ka0 = load_a16(kc, DMODEL, 0, d0);
    v16h ka1 = load_a16(kc, DMODEL, 0, d0 + 32);
    v16h ka2 = load_a16(kc, DMODEL, 16, d0);
    v16h ka3 = load_a16(kc, DMODEL, 16, d0 + 32);
    v16h va0 = load_a16(vc, KCHUNK, d0, 0);
    v16h va1 = load_a16(vc, KCHUNK, d0 + 16, 0);
    v16h va2 = load_a16(vc, KCHUNK, d0 + 32, 0);
    v16h va3 = load_a16(vc, KCHUNK, d0 + 48, 0);

    // S^T subtiles: 16 keys x 16 queries each
    v8f st0 = wmma16(ka0, qb0, zero8());
    st0 = wmma16(ka1, qb1, st0);
    v8f st1 = wmma16(ka2, qb0, zero8());
    st1 = wmma16(ka3, qb1, st1);

    // online softmax: per-lane chunk max over 16 regs + pair exchange
    float cmax = -INFINITY;
#pragma unroll
    for (int r = 0; r < 8; ++r) {
      cmax = fmaxf(cmax, st0[r]);
      cmax = fmaxf(cmax, st1[r]);
    }
    cmax = fmaxf(cmax, __shfl_xor(cmax, 16, 32));
    float mnew = fmaxf(mrow, cmax);
    float fac = __builtin_amdgcn_exp2f(mrow - mnew);
    mrow = mnew;

    float p0[8], p1[8];
    float csum = 0.0f;
#pragma unroll
    for (int r = 0; r < 8; ++r) {
      p0[r] = __builtin_amdgcn_exp2f(st0[r] - mrow);
      p1[r] = __builtin_amdgcn_exp2f(st1[r] - mrow);
      csum += p0[r] + p1[r];
    }
    csum += __shfl_xor(csum, 16, 32);
    lrow = lrow * fac + csum;

#pragma unroll
    for (int t = 0; t < 4; ++t)
#pragma unroll
      for (int i = 0; i < 8; ++i) oacc[t][i] *= fac;

    // build P^T B-fragment (32 keys x 16 queries) from C-layout tiles:
    // one shfl_xor(16) per register swaps the halves that live cross-lane
    union { v16h v; f16 e[16]; } pb;
#pragma unroll
    for (int r = 0; r < 8; ++r) {
      float t = __shfl_xor(s ? p0[r] : p1[r], 16, 32);
      float lo = s ? t : p0[r];
      float hi = s ? p1[r] : t;
      pb.e[r] = (f16)lo;
      pb.e[8 + r] = (f16)hi;
    }

    // O^T += V^T(16d x 32k) * P^T(32k x 16q), 4 d-tiles
    oacc[0] = wmma16(va0, pb.v, oacc[0]);
    oacc[1] = wmma16(va1, pb.v, oacc[1]);
    oacc[2] = wmma16(va2, pb.v, oacc[2]);
    oacc[3] = wmma16(va3, pb.v, oacc[3]);
    __syncthreads();            // all heads done reading before buffer reuse
  }

  // normalize and store AO row-major (coalesced b128 per lane per tile)
  float inv = 1.0f / lrow;
  int q = q0 + (l & 15);
#pragma unroll
  for (int dt = 0; dt < 4; ++dt) {
    v8h o;
#pragma unroll
    for (int r = 0; r < 8; ++r) o[r] = (f16)(oacc[dt][r] * inv);
    *(v8h*)(AOh + q * DMODEL + d0 + dt * 16 + 8 * s) = o;
  }
}

// ---------------- phase 3: output projection (fp32 out) ----------------
__global__ void __launch_bounds__(128) k_proj_out(
    const f16* __restrict__ AOh, const f16* __restrict__ WoT,
    const float* __restrict__ bo, float* __restrict__ out) {
  int wid = threadIdx.x >> 5;
  int m0 = (blockIdx.x * 2 + (wid >> 1)) * 16;
  int nbase = (wid & 1) * 128;

  v8f acc[8];
#pragma unroll
  for (int t = 0; t < 8; ++t) acc[t] = zero8();

  v16h a = load_a16(AOh, DMODEL, m0, 0);
  for (int k0 = 0; k0 < DMODEL; k0 += 32) {
    v16h an = (k0 + 32 < DMODEL) ? load_a16(AOh, DMODEL, m0, k0 + 32) : a;
    v16h b0 = load_b16(WoT, DMODEL, k0, nbase);
    v16h b1 = load_b16(WoT, DMODEL, k0, nbase + 16);
#pragma unroll
    for (int nt = 0; nt < 8; ++nt) {
      v16h bn = (nt + 2 < 8) ? load_b16(WoT, DMODEL, k0, nbase + (nt + 2) * 16) : b0;
      acc[nt] = wmma16(a, b0, acc[nt]);
      b0 = b1;
      b1 = bn;
    }
    a = an;
  }

  int l = threadIdx.x & 31;
  int s = l >> 4;
#pragma unroll
  for (int nt = 0; nt < 8; ++nt) {
    int n = nbase + nt * 16 + (l & 15);
    float b = bo[n];
#pragma unroll
    for (int r = 0; r < 8; ++r)
      out[(m0 + r + 8 * s) * DMODEL + n] = acc[nt][r] + b;
  }
}

extern "C" void kernel_launch(void* const* d_in, const int* in_sizes, int n_in,
                              void* d_out, int out_size, void* d_ws, size_t ws_size,
                              hipStream_t stream) {
  const float* x  = (const float*)d_in[0];
  const float* Wq = (const float*)d_in[1];
  const float* bq = (const float*)d_in[2];
  const float* Wk = (const float*)d_in[3];
  const float* bk = (const float*)d_in[4];
  const float* Wv = (const float*)d_in[5];
  const float* bv = (const float*)d_in[6];
  const float* Wo = (const float*)d_in[7];
  const float* bo = (const float*)d_in[8];
  float* out = (float*)d_out;

  // workspace layout (bytes); requires ~20.5 MB
  char* w = (char*)d_ws;
  const size_t MB = 1024 * 1024;
  f16* xh  = (f16*)(w + 0 * MB);   // 8192x256 f16  (4 MB)
  f16* Qh  = (f16*)(w + 4 * MB);   // 8192x256 f16  (4 MB), scale*log2e folded
  f16* Kh  = (f16*)(w + 8 * MB);   // 8192x256 f16  (4 MB)
  f16* VTh = (f16*)(w + 12 * MB);  // 256x8192 f16  (4 MB), transposed V
  f16* AOh = (f16*)(w + 16 * MB);  // 8192x256 f16  (4 MB)
  f16* WqT = (f16*)(w + 20 * MB);  // 4 x 256x256 f16 (512 KB)
  f16* WkT = WqT + DMODEL * DMODEL;
  f16* WvT = WkT + DMODEL * DMODEL;
  f16* WoT = WvT + DMODEL * DMODEL;

  // prep
  k_cvt_x<<<(N_TOK * DMODEL) / (256 * 8), 256, 0, stream>>>(x, xh);
  k_prep_w<<<dim3(16, 16, 4), dim3(16, 16), 0, stream>>>(Wq, Wk, Wv, Wo,
                                                         WqT, WkT, WvT, WoT);
  // QKV projection: 512 row-tiles x 2 column halves x 3 matrices
  k_proj_qkv<<<dim3(N_TOK / 32, 3), 128, 0, stream>>>(xh, WqT, WkT, WvT,
                                                      bq, bk, bv, Qh, Kh, VTh);
  // flash attention: 512 query-tiles, 4 heads (one wave each), TDM-fed
  k_attn<<<N_TOK / 16, 128, 0, stream>>>(Qh, Kh, VTh, AOh);
  // output projection
  k_proj_out<<<N_TOK / 32, 128, 0, stream>>>(AOh, WoT, bo, out);
}
